// SelfAttention_25735444037718
// MI455X (gfx1250) — matrix operand order repaired and checked by reading, not compiled, not run on hardware
//
#include <hip/hip_runtime.h>
#include <hip/hip_bf16.h>

// ---------------------------------------------------------------------------
// SelfAttention (channel attention) for MI455X / gfx1250, wave32 + WMMA bf16.
//
//   P  = [Wq;Wk;Wv] @ x + b           (K1,  bf16 WMMA, [192,512]x[512,4096]/batch)
//   A  = softmax(Q @ reshape(K))      (K2a, VALU, tiny)
//   M  = Wo @ A                       (K2b, VALU, tiny)
//   y  = gamma*(M @ V + bo) + x       (K3,  bf16 WMMA, [512,64]x[64,4096]/batch)
//
// LDS pitch = 72 bf16 (144 B): every WMMA fragment base is 16-B aligned so
// each fragment is exactly two ds_load_b128, and 144 B = 36 dwords gives a
// conflict-free bank tiling across the 16 lanes of a fragment row/col group.
// ---------------------------------------------------------------------------

typedef __attribute__((ext_vector_type(16))) __bf16 v16bf;
typedef __attribute__((ext_vector_type(8)))  __bf16 v8bf;
typedef __attribute__((ext_vector_type(8)))  float  v8f;

#define WMMA_BF16(a, b, c) \
    __builtin_amdgcn_wmma_f32_16x16x32_bf16(false, (a), false, (b), (short)0, (c), false, false)

static constexpr int Bz   = 16;
static constexpr int C    = 512;
static constexpr int N    = 4096;
static constexpr int c64  = 64;
static constexpr int MR   = 192;     // rows of fused QKV output (Q|K|V)
static constexpr int PIT  = 72;      // LDS pitch in bf16 elems (144 B, 16-B aligned)

// workspace layout (bytes)
static constexpr size_t P_BYTES    = (size_t)Bz * MR * N * 2;          // 25,165,824
static constexpr size_t ATTN_OFF   = P_BYTES;
static constexpr size_t ATTN_BYTES = (size_t)Bz * c64 * c64 * 4;       // 262,144
static constexpr size_t M_OFF      = ATTN_OFF + ATTN_BYTES;

__device__ __forceinline__ __bf16 to_bf16(float f) { return (__bf16)f; }

__device__ __forceinline__ unsigned pack2(float a, float b) {
    unsigned short u0 = __builtin_bit_cast(unsigned short, to_bf16(a));
    unsigned short u1 = __builtin_bit_cast(unsigned short, to_bf16(b));
    return (unsigned)u0 | ((unsigned)u1 << 16);
}

// A fragment (16x32 bf16): lane holds row, K = kb+{0..7} and kb+{16..23}.
__device__ __forceinline__ v16bf load_a_frag(const __bf16* base16 /*16B aligned*/) {
    const v8bf* p = (const v8bf*)base16;
    v8bf lo = p[0];                 // K +0..7
    v8bf hi = p[2];                 // K +16..23
    return __builtin_shufflevector(lo, hi, 0,1,2,3,4,5,6,7,8,9,10,11,12,13,14,15);
}
// B fragment (32x16 bf16): lane holds col, K = kb+{0..7} and kb+{16..23}.
__device__ __forceinline__ v16bf load_b_frag(const __bf16* base16 /*16B aligned*/) {
    const v8bf* p = (const v8bf*)base16;
    v8bf lo = p[0];
    v8bf hi = p[2];
    return __builtin_shufflevector(lo, hi, 0,1,2,3,4,5,6,7,8,9,10,11,12,13,14,15);
}

// ===========================================================================
// Kernel 1: P[b, 0:192, n] = [Wq;Wk;Wv] @ x[b] + [bq;bk;bv]   (bf16 out)
// grid (N/128, B), 256 threads = 8 waves; wave w owns the 16-col strip w.
// ===========================================================================
__global__ __launch_bounds__(256) void qkv_gemm_kernel(
    const float* __restrict__ x,
    const float* __restrict__ Wq, const float* __restrict__ bq,
    const float* __restrict__ Wk, const float* __restrict__ bk,
    const float* __restrict__ Wv, const float* __restrict__ bv,
    __bf16* __restrict__ P)
{
    __shared__ alignas(16) __bf16 lA[MR  * PIT];  // weights [row][k]  27.0 KB
    __shared__ alignas(16) __bf16 lB[128 * PIT];  // x tile  [n][k]    18.0 KB
    __shared__ float lbias[MR];

    const int b    = blockIdx.y;
    const int n0   = blockIdx.x * 128;
    const int tid  = threadIdx.x;
    const int lane = tid & 31;
    const int wave = tid >> 5;

    if (tid < MR) {
        const int m = tid;
        lbias[m] = (m < 64) ? bq[m] : (m < 128) ? bk[m - 64] : bv[m - 128];
    }

    const float* xb = x + (size_t)b * C * N;

    v8f acc[12];
    #pragma unroll
    for (int i = 0; i < 12; ++i)
        #pragma unroll
        for (int e = 0; e < 8; ++e) acc[i][e] = 0.0f;

    for (int k0 = 0; k0 < C; k0 += 64) {
        // ---- stage A: 192 x 64 weight chunk, packed 2xbf16 per u32 write ----
        for (int j = 0; j < 24; ++j) {
            const int i   = tid + j * 256;         // 0..6143 (pairs)
            const int m   = i >> 5;
            const int kk2 = (i & 31) * 2;
            const float* Wsrc = (m < 64)  ? (Wq + (size_t)m * C)
                              : (m < 128) ? (Wk + (size_t)(m - 64) * C)
                                          : (Wv + (size_t)(m - 128) * C);
            *(unsigned*)&lA[m * PIT + kk2] = pack2(Wsrc[k0 + kk2], Wsrc[k0 + kk2 + 1]);
        }
        // ---- stage B: x tile [64 k][128 n] -> lB[n][k], packed k-pairs ----
        for (int j = 0; j < 16; ++j) {
            const int i   = tid + j * 256;         // 0..4095 (pairs)
            const int kk2 = (i >> 7) * 2;
            const int nn  = i & 127;
            const size_t g = (size_t)(k0 + kk2) * N + n0 + nn;
            *(unsigned*)&lB[nn * PIT + kk2] = pack2(xb[g], xb[g + N]);
        }
        // prefetch next x k-chunk (speculative; global_prefetch_b8)
        if (k0 + 64 < C)
            __builtin_prefetch(&xb[(size_t)(k0 + 64) * N + n0 + (tid & 127)], 0, 1);
        __syncthreads();

        #pragma unroll
        for (int kc = 0; kc < 64; kc += 32) {
            const v16bf bfrag = load_b_frag(
                &lB[(wave * 16 + (lane & 15)) * PIT + kc + ((lane & 16) ? 8 : 0)]);
            const int akb = kc + ((lane & 16) ? 8 : 0);
            #pragma unroll
            for (int mt = 0; mt < 12; ++mt) {
                const v16bf afrag = load_a_frag(&lA[(mt * 16 + (lane & 15)) * PIT + akb]);
                acc[mt] = WMMA_BF16(afrag, bfrag, acc[mt]);
            }
        }
        __syncthreads();
    }

    // ---- epilogue: + bias, store bf16 ----
    __bf16* Pb = P + (size_t)b * MR * N;
    const int col   = n0 + wave * 16 + (lane & 15);
    const int rbase = (lane & 16) ? 8 : 0;
    #pragma unroll
    for (int mt = 0; mt < 12; ++mt) {
        #pragma unroll
        for (int r = 0; r < 8; ++r) {
            const int row = mt * 16 + rbase + r;
            Pb[(size_t)row * N + col] = to_bf16(acc[mt][r] + lbias[row]);
        }
    }
}

// ===========================================================================
// Kernel 2a: attn[b,q,:] = softmax_d( sum_n Q[q,n] * Kflat[n*64+d] )
// grid (B, 4): each block owns 16 q-rows; 256 thr = 16 q x 16 d-groups(x4).
// ===========================================================================
__global__ __launch_bounds__(256) void attn_softmax_kernel(
    const __bf16* __restrict__ P, float* __restrict__ attn)
{
    __shared__ __bf16 lK[128 * 64];   // Keys chunk [n][d]  16 KB
    __shared__ __bf16 lQ[16 * 128];   // Q chunk    [q][n]   4 KB
    __shared__ float  lg[16 * 64];    // logits              4 KB

    const int b   = blockIdx.x;
    const int qs  = blockIdx.y;       // 0..3
    const int tid = threadIdx.x;
    const int q   = tid >> 4;         // 0..15
    const int dg  = tid & 15;         // d in [dg*4, dg*4+4)

    const __bf16* Qg = P + (size_t)b * MR * N + (size_t)(qs * 16) * N;
    const __bf16* Kg = P + (size_t)b * MR * N + (size_t)64 * N;   // K rows, linear

    float p0 = 0.f, p1 = 0.f, p2 = 0.f, p3 = 0.f;

    for (int n0 = 0; n0 < N; n0 += 128) {
        for (int j = 0; j < 32; ++j) {            // Keys chunk: identity copy
            const int i = tid + j * 256;          // 0..8191
            lK[i] = Kg[(size_t)n0 * 64 + i];
        }
        for (int j = 0; j < 8; ++j) {             // Q chunk: 16 rows x 128
            const int i  = tid + j * 256;         // 0..2047
            const int qq = i >> 7, nn = i & 127;
            lQ[qq * 128 + nn] = Qg[(size_t)qq * N + n0 + nn];
        }
        __syncthreads();
        for (int n = 0; n < 128; ++n) {
            const float qv = (float)lQ[q * 128 + n];
            const __bf16* kr = &lK[n * 64 + dg * 4];
            p0 += qv * (float)kr[0];
            p1 += qv * (float)kr[1];
            p2 += qv * (float)kr[2];
            p3 += qv * (float)kr[3];
        }
        __syncthreads();
    }

    lg[q * 64 + dg * 4 + 0] = p0;
    lg[q * 64 + dg * 4 + 1] = p1;
    lg[q * 64 + dg * 4 + 2] = p2;
    lg[q * 64 + dg * 4 + 3] = p3;
    __syncthreads();

    if (tid < 16) {                               // row-wise softmax (64 wide)
        float mx = -1e30f;
        for (int d = 0; d < 64; ++d) mx = fmaxf(mx, lg[tid * 64 + d]);
        float s = 0.f;
        for (int d = 0; d < 64; ++d) { float e = __expf(lg[tid * 64 + d] - mx); lg[tid * 64 + d] = e; s += e; }
        const float inv = 1.0f / s;
        for (int d = 0; d < 64; ++d) lg[tid * 64 + d] *= inv;
    }
    __syncthreads();

    for (int j = 0; j < 4; ++j) {
        const int i  = tid + j * 256;             // 0..1023
        const int qq = i >> 6, d = i & 63;
        attn[(size_t)b * c64 * c64 + (size_t)(qs * 16 + qq) * 64 + d] = lg[i];
    }
}

// ===========================================================================
// Kernel 2b: M[b] = Wo @ attn[b]   ([512,64]x[64,64], bf16 out)
// grid (B, 8): block owns 64 o-rows; thread = (o = os*64 + tid/4, 16 d's).
// ===========================================================================
__global__ __launch_bounds__(256) void proj_attn_kernel(
    const float* __restrict__ Wo, const float* __restrict__ attn,
    __bf16* __restrict__ Mm)
{
    __shared__ float lattn[64 * 64];              // 16 KB

    const int b   = blockIdx.x;
    const int os  = blockIdx.y;                   // 0..7
    const int tid = threadIdx.x;

    for (int j = 0; j < 16; ++j)
        lattn[tid + j * 256] = attn[(size_t)b * c64 * c64 + tid + j * 256];
    __syncthreads();

    const int o  = os * 64 + (tid >> 2);
    const int dg = tid & 3;                       // d in [dg*16, dg*16+16)
    const float* wrow = Wo + (size_t)o * 64;

    float acc[16];
    #pragma unroll
    for (int j = 0; j < 16; ++j) acc[j] = 0.f;

    for (int cc = 0; cc < 64; ++cc) {
        const float w = wrow[cc];
        const float* ar = &lattn[cc * 64 + dg * 16];
        #pragma unroll
        for (int j = 0; j < 16; ++j) acc[j] += w * ar[j];
    }

    __bf16* mrow = Mm + (size_t)b * C * c64 + (size_t)o * 64 + dg * 16;
    #pragma unroll
    for (int j = 0; j < 16; ++j) mrow[j] = to_bf16(acc[j]);
}

// ===========================================================================
// Kernel 3: out[b] = gamma*(M[b] @ V[b] + bo) + x[b]    (WMMA, K=64)
// grid (N/64, 2, B), 256 thr = 8 waves; wave = (mq 0..3) x (nq 0..1),
// block tile 256 rows x 64 cols; each wave 4x2 WMMA accum tiles.
// ===========================================================================
__global__ __launch_bounds__(256) void out_gemm_kernel(
    const __bf16* __restrict__ Mm, const __bf16* __restrict__ P,
    const float* __restrict__ x,  const float* __restrict__ bo,
    const float* __restrict__ gamma_p, float* __restrict__ out)
{
    __shared__ alignas(16) __bf16 lM[256 * PIT];  // M rows [r][k]  36.0 KB
    __shared__ alignas(16) __bf16 lV[64  * PIT];  // V tile [n][k]   9.0 KB
    __shared__ float lbo[256];

    const int b    = blockIdx.z;
    const int mblk = blockIdx.y;                  // 0..1 (256 rows each)
    const int n0   = blockIdx.x * 64;
    const int tid  = threadIdx.x;
    const int lane = tid & 31;
    const int wave = tid >> 5;
    const int mq   = wave >> 1;                   // 0..3 (64 rows)
    const int nq   = wave & 1;                    // 0..1 (32 cols)

    // stage M (256x64 bf16): copy u32 pairs straight through
    const unsigned* Mg32 = (const unsigned*)(Mm + (size_t)b * C * c64 + (size_t)mblk * 256 * 64);
    for (int j = 0; j < 32; ++j) {
        const int i  = tid + j * 256;             // 0..8191 (u32 pairs)
        const int r  = i >> 5;
        const int k2 = (i & 31) * 2;
        *(unsigned*)&lM[r * PIT + k2] = Mg32[i];
    }
    // stage V transposed -> [n][k], packed k-pairs
    const __bf16* Vg = P + (size_t)b * MR * N + (size_t)128 * N;
    for (int j = 0; j < 8; ++j) {
        const int i  = tid + j * 256;             // 0..2047 (pairs)
        const int d2 = (i >> 6) * 2;
        const int nn = i & 63;
        const size_t g = (size_t)d2 * N + n0 + nn;
        unsigned short u0 = __builtin_bit_cast(unsigned short, Vg[g]);
        unsigned short u1 = __builtin_bit_cast(unsigned short, Vg[g + N]);
        *(unsigned*)&lV[nn * PIT + d2] = (unsigned)u0 | ((unsigned)u1 << 16);
    }
    lbo[tid] = bo[mblk * 256 + tid];
    __syncthreads();

    v8f acc[4][2];
    #pragma unroll
    for (int mt = 0; mt < 4; ++mt)
        #pragma unroll
        for (int nt = 0; nt < 2; ++nt)
            #pragma unroll
            for (int e = 0; e < 8; ++e) acc[mt][nt][e] = 0.0f;

    #pragma unroll
    for (int kc = 0; kc < 64; kc += 32) {
        v16bf bfrag[2];
        #pragma unroll
        for (int nt = 0; nt < 2; ++nt)
            bfrag[nt] = load_b_frag(
                &lV[(nq * 32 + nt * 16 + (lane & 15)) * PIT + kc + ((lane & 16) ? 8 : 0)]);
        const int akb = kc + ((lane & 16) ? 8 : 0);
        #pragma unroll
        for (int mt = 0; mt < 4; ++mt) {
            const v16bf afrag = load_a_frag(&lM[(mq * 64 + mt * 16 + (lane & 15)) * PIT + akb]);
            #pragma unroll
            for (int nt = 0; nt < 2; ++nt)
                acc[mt][nt] = WMMA_BF16(afrag, bfrag[nt], acc[mt][nt]);
        }
    }

    // ---- fused epilogue: gamma*(acc + bo) + x ----
    const float g = gamma_p[0];
    const float* xb = x   + (size_t)b * C * N;
    float*       ob = out + (size_t)b * C * N;
    const int rbase = (lane & 16) ? 8 : 0;
    #pragma unroll
    for (int mt = 0; mt < 4; ++mt) {
        #pragma unroll
        for (int nt = 0; nt < 2; ++nt) {
            const int col = n0 + nq * 32 + nt * 16 + (lane & 15);
            #pragma unroll
            for (int r = 0; r < 8; ++r) {
                const int rloc = mq * 64 + mt * 16 + rbase + r;
                const size_t idx = (size_t)(mblk * 256 + rloc) * N + col;
                ob[idx] = g * (acc[mt][nt][r] + lbo[rloc]) + xb[idx];
            }
        }
    }
}

// ===========================================================================
extern "C" void kernel_launch(void* const* d_in, const int* in_sizes, int n_in,
                              void* d_out, int out_size, void* d_ws, size_t ws_size,
                              hipStream_t stream) {
    (void)in_sizes; (void)n_in; (void)out_size; (void)ws_size;

    const float* x     = (const float*)d_in[0];
    const float* Wq    = (const float*)d_in[1];
    const float* bq    = (const float*)d_in[2];
    const float* Wk    = (const float*)d_in[3];
    const float* bk    = (const float*)d_in[4];
    const float* Wv    = (const float*)d_in[5];
    const float* bv    = (const float*)d_in[6];
    const float* Wo    = (const float*)d_in[7];
    const float* bo    = (const float*)d_in[8];
    const float* gamma = (const float*)d_in[9];

    __bf16* P    = (__bf16*)d_ws;
    float*  attn = (float*)((char*)d_ws + ATTN_OFF);
    __bf16* Mm   = (__bf16*)((char*)d_ws + M_OFF);
    float*  outp = (float*)d_out;

    qkv_gemm_kernel    <<<dim3(N / 128, Bz),    256, 0, stream>>>(x, Wq, bq, Wk, bk, Wv, bv, P);
    attn_softmax_kernel<<<dim3(Bz, 4),          256, 0, stream>>>(P, attn);
    proj_attn_kernel   <<<dim3(Bz, 8),          256, 0, stream>>>(Wo, attn, Mm);
    out_gemm_kernel    <<<dim3(N / 64, 2, Bz),  256, 0, stream>>>(Mm, P, x, bo, gamma, outp);
}
